// RGCN_62079457296840
// MI455X (gfx1250) — compile-verified, matrix operand order checked by reading, not run on hardware
//
#include <hip/hip_runtime.h>

#define NNODES 50000
#define NEDGES 500000
#define EMB 100
#define HID 128
#define NB  4

typedef __attribute__((ext_vector_type(16))) _Float16 v16h;
typedef __attribute__((ext_vector_type(8)))  float    v8f;

// ------------------------------ helpers ------------------------------------
__device__ __forceinline__ float warpSum32(float v) {
#pragma unroll
  for (int m = 16; m >= 1; m >>= 1) v += __shfl_xor(v, m, 32);
  return v;
}

__device__ __forceinline__ unsigned fkey(float f) {
  unsigned u = __float_as_uint(f);
  return (u & 0x80000000u) ? ~u : (u | 0x80000000u);   // monotone float->uint
}

__device__ __forceinline__ float softplusf(float x) {
  return log1pf(expf(-fabsf(x))) + fmaxf(x, 0.0f);
}

// ------------------------------ init ---------------------------------------
__global__ void k_zero(unsigned* __restrict__ p, long n) {
  long i = (long)blockIdx.x * blockDim.x + threadIdx.x;
  long st = (long)gridDim.x * blockDim.x;
  for (; i < n; i += st) p[i] = 0u;
}

// x = relu(emb + ebias), cast to f16, K padded 100 -> 128 with zeros
__global__ void k_prep_x(const float* __restrict__ emb, const float* __restrict__ ebias,
                         _Float16* __restrict__ x1h) {
  long n = (long)NNODES * 128;
  long i = (long)blockIdx.x * blockDim.x + threadIdx.x;
  long st = (long)gridDim.x * blockDim.x;
  for (; i < n; i += st) {
    int k = (int)(i & 127);
    long nd = i >> 7;
    float v = 0.0f;
    if (k < EMB) v = fmaxf(emb[nd * EMB + k] + ebias[k], 0.0f);
    x1h[i] = (_Float16)v;
  }
}

// Wt1[c][k] (640 x 128, k contiguous): c<512 -> basis1[b=c>>7][k][o=c&127]; c>=512 -> root1[k][c-512]
__global__ void k_prep_w1(const float* __restrict__ basis1, const float* __restrict__ root1,
                          _Float16* __restrict__ Wt1) {
  int i = blockIdx.x * blockDim.x + threadIdx.x;
  if (i >= 640 * 128) return;
  int k = i & 127, c = i >> 7;
  float v = 0.0f;
  if (k < EMB) {
    if (c < NB * HID) { int b = c >> 7, o = c & 127; v = basis1[((long)b * EMB + k) * HID + o]; }
    else              { int o = c - NB * HID;        v = root1[(long)k * HID + o]; }
  }
  Wt1[i] = (_Float16)v;
}

// Wt2[c][k] (512 x 128): c<400 -> basis2[b=c/100][k][c%100]; 400<=c<500 -> root2[k][c-400]; else 0
__global__ void k_prep_w2(const float* __restrict__ basis2, const float* __restrict__ root2,
                          _Float16* __restrict__ Wt2) {
  int i = blockIdx.x * blockDim.x + threadIdx.x;
  if (i >= 512 * 128) return;
  int k = i & 127, c = i >> 7;
  float v = 0.0f;
  if (c < NB * EMB)            { int b = c / EMB, o = c % EMB; v = basis2[((long)b * HID + k) * EMB + o]; }
  else if (c < NB * EMB + EMB) { int o = c - NB * EMB;         v = root2[(long)k * EMB + o]; }
  Wt2[i] = (_Float16)v;
}

// --------------------------- WMMA GEMM -------------------------------------
// C[M x NCOLS] = A[M x 128]f16 * Bt[NCOLS x 128]f16^T.
// Each wave computes a 16x64 strip (1 M-tile x 4 N-tiles): the A fragment is
// loaded once per K-block and reused by 4 back-to-back v_wmma issues (no
// D->A/B overlap, so no hazard NOPs). Fragment fills follow the CDNA5 16-bit
// layout: lane = row, half = lane>>4 selects the K-group; each lane's data is
// two contiguous 16B chunks -> global_load_b128 pairs.
template <int NCOLS>
__global__ __launch_bounds__(256) void k_wmma_gemm(const _Float16* __restrict__ A,
                                                   const _Float16* __restrict__ Bt,
                                                   float* __restrict__ C,
                                                   int totalWaveTiles) {
  constexpr int NT4 = NCOLS / 64;                       // N-groups of 4 tiles
  int t = (int)((blockIdx.x * blockDim.x + threadIdx.x) >> 5);
  if (t >= totalWaveTiles) return;                      // wave-uniform exit
  int lane = threadIdx.x & 31;
  int mTile = t / NT4, nGrp = t % NT4;
  int half = lane >> 4, rlo = lane & 15;

  const _Float16* ap = A  + (size_t)(mTile * 16 + rlo) * 128;
  const _Float16* bp = Bt + (size_t)(nGrp * 64 + rlo) * 128;

  v8f acc[4];
#pragma unroll
  for (int j = 0; j < 4; ++j) acc[j] = (v8f){0.f, 0.f, 0.f, 0.f, 0.f, 0.f, 0.f, 0.f};

#pragma unroll
  for (int kb = 0; kb < 128; kb += 32) {
    v16h a;
#pragma unroll
    for (int v = 0; v < 8; ++v) {
      int k = kb + ((v < 4) ? (2 * v + 8 * half) : (16 + 2 * (v - 4) + 8 * half));
      a[2 * v]     = ap[k];
      a[2 * v + 1] = ap[k + 1];
    }
#pragma unroll
    for (int j = 0; j < 4; ++j) {
      const _Float16* bj = bp + (size_t)j * 16 * 128;
      v16h b;
#pragma unroll
      for (int v = 0; v < 8; ++v) {
        int k = kb + ((v < 4) ? (2 * v + 8 * half) : (16 + 2 * (v - 4) + 8 * half));
        b[2 * v]     = bj[k];
        b[2 * v + 1] = bj[k + 1];
      }
      acc[j] = __builtin_amdgcn_wmma_f32_16x16x32_f16(false, a, false, b, (short)0,
                                                      acc[j], false, false);
    }
  }

#pragma unroll
  for (int j = 0; j < 4; ++j) {
    float* cp = C + (size_t)(mTile * 16) * NCOLS + nGrp * 64 + j * 16 + rlo;
#pragma unroll
    for (int v = 0; v < 8; ++v) cp[(size_t)(v + 8 * half) * NCOLS] = acc[j][v];
  }
}

// --------------------------- edge kernels ----------------------------------
__global__ void k_msg1(const int* __restrict__ src, const int* __restrict__ dst,
                       const int* __restrict__ et, const float* __restrict__ comp1,
                       const float* __restrict__ xb1, float* __restrict__ accA,
                       float* __restrict__ deg) {
  int warp = (blockIdx.x * blockDim.x + threadIdx.x) >> 5;
  int lane = threadIdx.x & 31;
  int nW = (gridDim.x * blockDim.x) >> 5;
  for (int e = warp; e < NEDGES; e += nW) {
    int s = src[e], d = dst[e], r = et[e];
    float c0 = comp1[r * 4 + 0], c1 = comp1[r * 4 + 1];
    float c2 = comp1[r * 4 + 2], c3 = comp1[r * 4 + 3];
    const float* xs = xb1 + (size_t)s * 640;
    int en = e + nW;
    if (en < NEDGES) __builtin_prefetch(xb1 + (size_t)src[en] * 640 + lane * 16, 0, 1);
#pragma unroll
    for (int j = 0; j < 4; ++j) {
      int o = lane + 32 * j;
      float m = c0 * xs[o] + c1 * xs[128 + o] + c2 * xs[256 + o] + c3 * xs[384 + o];
      atomicAdd(&accA[(size_t)d * 128 + o], m);
    }
    if (lane == 0) atomicAdd(&deg[d], 1.0f);
  }
}

__global__ void k_msg2(const int* __restrict__ src, const int* __restrict__ dst,
                       const int* __restrict__ et, const float* __restrict__ comp2,
                       const float* __restrict__ xb2, float* __restrict__ accZ) {
  int warp = (blockIdx.x * blockDim.x + threadIdx.x) >> 5;
  int lane = threadIdx.x & 31;
  int nW = (gridDim.x * blockDim.x) >> 5;
  for (int e = warp; e < NEDGES; e += nW) {
    int s = src[e], d = dst[e], r = et[e];
    float c0 = comp2[r * 4 + 0], c1 = comp2[r * 4 + 1];
    float c2 = comp2[r * 4 + 2], c3 = comp2[r * 4 + 3];
    const float* xs = xb2 + (size_t)s * 512;
    int en = e + nW;
    if (en < NEDGES) __builtin_prefetch(xb2 + (size_t)src[en] * 512 + lane * 16, 0, 1);
#pragma unroll
    for (int j = 0; j < 4; ++j) {
      int o = lane + 32 * j;
      if (o < EMB) {
        float m = c0 * xs[o] + c1 * xs[100 + o] + c2 * xs[200 + o] + c3 * xs[300 + o];
        atomicAdd(&accZ[(size_t)d * EMB + o], m);
      }
    }
  }
}

// h1 = relu(agg/deg + x@root1 + bias1)  (root part lives in xb1 cols 512..639)
__global__ void k_fin1(const float* __restrict__ accA, const float* __restrict__ deg,
                       const float* __restrict__ xb1, const float* __restrict__ bias1,
                       _Float16* __restrict__ h1h) {
  long n = (long)NNODES * 128;
  long i = (long)blockIdx.x * blockDim.x + threadIdx.x;
  long st = (long)gridDim.x * blockDim.x;
  for (; i < n; i += st) {
    int o = (int)(i & 127);
    long nd = i >> 7;
    float dv = fmaxf(deg[nd], 1.0f);
    float h = accA[i] / dv + xb1[nd * 640 + 512 + o] + bias1[o];
    h1h[i] = (_Float16)fmaxf(h, 0.0f);
  }
}

// z = agg/deg + h@root2 + bias2  (root part lives in xb2 cols 400..499)
__global__ void k_fin2(const float* __restrict__ accZ, const float* __restrict__ deg,
                       const float* __restrict__ xb2, const float* __restrict__ bias2,
                       float* __restrict__ z) {
  long n = (long)NNODES * EMB;
  long i = (long)blockIdx.x * blockDim.x + threadIdx.x;
  long st = (long)gridDim.x * blockDim.x;
  for (; i < n; i += st) {
    long nd = i / EMB;
    int o = (int)(i % EMB);
    float dv = fmaxf(deg[nd], 1.0f);
    z[i] = accZ[i] / dv + xb2[nd * 512 + 400 + o] + bias2[o];
  }
}

// --------------------------- distmult / loss / auc -------------------------
__global__ void k_distmult(const int* __restrict__ src, const int* __restrict__ dst,
                           const int* __restrict__ et, const int* __restrict__ negt,
                           const float* __restrict__ z, const float* __restrict__ rel,
                           float* __restrict__ pos, float* __restrict__ neg) {
  int warp = (blockIdx.x * blockDim.x + threadIdx.x) >> 5;
  int lane = threadIdx.x & 31;
  int nW = (gridDim.x * blockDim.x) >> 5;
  for (int e = warp; e < NEDGES; e += nW) {
    int s = src[e], d = dst[e], r = et[e], q = negt[e];
    const float* zs = z + (size_t)s * EMB;
    const float* zd = z + (size_t)d * EMB;
    const float* zq = z + (size_t)q * EMB;
    const float* re = rel + (size_t)r * EMB;
    float p = 0.0f, ng = 0.0f;
    for (int o = lane; o < EMB; o += 32) {
      float a = zs[o] * re[o];
      p  += a * zd[o];
      ng += a * zq[o];
    }
    p = warpSum32(p);
    ng = warpSum32(ng);
    if (lane == 0) { pos[e] = p; neg[e] = ng; }
  }
}

__global__ void k_loss(const float* __restrict__ pos, const float* __restrict__ neg,
                       float* __restrict__ sums) {
  float sp = 0.0f, sn = 0.0f;
  long i = (long)blockIdx.x * blockDim.x + threadIdx.x;
  long st = (long)gridDim.x * blockDim.x;
  for (; i < NEDGES; i += st) {
    sp += softplusf(-pos[i]);
    sn += softplusf(neg[i]);
  }
  __shared__ float sA[256], sB[256];
  int tid = threadIdx.x;
  sA[tid] = sp; sB[tid] = sn;
  __syncthreads();
  for (int s = 128; s > 0; s >>= 1) {
    if (tid < s) { sA[tid] += sA[tid + s]; sB[tid] += sB[tid + s]; }
    __syncthreads();
  }
  if (tid == 0) { atomicAdd(&sums[0], sA[0]); atomicAdd(&sums[1], sB[0]); }
}

__global__ void k_hist(const float* __restrict__ neg, unsigned* __restrict__ hist) {
  long i = (long)blockIdx.x * blockDim.x + threadIdx.x;
  long st = (long)gridDim.x * blockDim.x;
  for (; i < NEDGES; i += st) atomicAdd(&hist[fkey(neg[i]) >> 16], 1u);
}

__global__ void k_scan1(const unsigned* __restrict__ hist, unsigned* __restrict__ cdf,
                        unsigned* __restrict__ bsum) {
  __shared__ unsigned sh[256];
  int i = blockIdx.x * 256 + threadIdx.x;
  unsigned v = hist[i];
  sh[threadIdx.x] = v;
  __syncthreads();
  for (int o = 1; o < 256; o <<= 1) {
    unsigned x = (threadIdx.x >= o) ? sh[threadIdx.x - o] : 0u;
    __syncthreads();
    sh[threadIdx.x] += x;
    __syncthreads();
  }
  cdf[i] = sh[threadIdx.x] - v;                 // exclusive within block
  if (threadIdx.x == 255) bsum[blockIdx.x] = sh[255];
}

__global__ void k_scan2(unsigned* __restrict__ bsum) {
  __shared__ unsigned sh[256];
  int t = threadIdx.x;
  unsigned v = bsum[t];
  sh[t] = v;
  __syncthreads();
  for (int o = 1; o < 256; o <<= 1) {
    unsigned x = (t >= o) ? sh[t - o] : 0u;
    __syncthreads();
    sh[t] += x;
    __syncthreads();
  }
  bsum[t] = sh[t] - v;                          // exclusive
}

__global__ void k_scan3(unsigned* __restrict__ cdf, const unsigned* __restrict__ bsum) {
  cdf[blockIdx.x * 256 + threadIdx.x] += bsum[blockIdx.x];
}

// Mann-Whitney pair count: for each positive, #neg strictly below + 0.5*ties(bin)
__global__ void k_pairs(const float* __restrict__ pos, const unsigned* __restrict__ hist,
                        const unsigned* __restrict__ cdf, double* __restrict__ pairSum) {
  double loc = 0.0;
  long i = (long)blockIdx.x * blockDim.x + threadIdx.x;
  long st = (long)gridDim.x * blockDim.x;
  for (; i < NEDGES; i += st) {
    unsigned k = fkey(pos[i]) >> 16;
    loc += (double)cdf[k] + 0.5 * (double)hist[k];
  }
  __shared__ double sh[256];
  int tid = threadIdx.x;
  sh[tid] = loc;
  __syncthreads();
  for (int s = 128; s > 0; s >>= 1) {
    if (tid < s) sh[tid] += sh[tid + s];
    __syncthreads();
  }
  if (tid == 0) atomicAdd(pairSum, sh[0]);
}

__global__ void k_final(const float* __restrict__ sums, const double* __restrict__ pairSum,
                        float* __restrict__ outScal) {
  if (blockIdx.x == 0 && threadIdx.x == 0) {
    outScal[0] = 0.5f * (sums[0] + sums[1]) / (float)NEDGES;
    outScal[1] = (float)(pairSum[0] / ((double)NEDGES * (double)NEDGES));
  }
}

// ------------------------------ launch -------------------------------------
extern "C" void kernel_launch(void* const* d_in, const int* in_sizes, int n_in,
                              void* d_out, int out_size, void* d_ws, size_t ws_size,
                              hipStream_t stream) {
  const int*   edge_index = (const int*)d_in[0];
  const int*   src   = edge_index;
  const int*   dst   = edge_index + NEDGES;
  const int*   et    = (const int*)d_in[1];
  const int*   negt  = (const int*)d_in[2];
  const float* emb   = (const float*)d_in[3];
  const float* ebias = (const float*)d_in[4];
  const float* basis1= (const float*)d_in[5];
  const float* comp1 = (const float*)d_in[6];
  const float* root1 = (const float*)d_in[7];
  const float* bias1 = (const float*)d_in[8];
  const float* basis2= (const float*)d_in[9];
  const float* comp2 = (const float*)d_in[10];
  const float* root2 = (const float*)d_in[11];
  const float* bias2 = (const float*)d_in[12];
  const float* rel   = (const float*)d_in[13];

  char* ws = (char*)d_ws;
  size_t off = 0;
  auto take = [&](size_t bytes) -> void* {
    void* p = ws + off;
    off += (bytes + 255) & ~(size_t)255;
    return p;
  };
  float*    xb1  = (float*)   take((size_t)NNODES * 640 * 4);  // basis1(512)+root1(128) cols
  float*    xb2  = (float*)   take((size_t)NNODES * 512 * 4);  // basis2(400)+root2(100)+pad
  _Float16* x1h  = (_Float16*)take((size_t)NNODES * 128 * 2);
  _Float16* h1h  = (_Float16*)take((size_t)NNODES * 128 * 2);
  float*    z    = (float*)   take((size_t)NNODES * EMB * 4);
  _Float16* Wt1  = (_Float16*)take((size_t)640 * 128 * 2);
  _Float16* Wt2  = (_Float16*)take((size_t)512 * 128 * 2);
  float*    negb = (float*)   take((size_t)NEDGES * 4);
  size_t zeroOff = off;                                        // ---- zeroed region ----
  float*    accA = (float*)   take((size_t)NNODES * 128 * 4);
  float*    accZ = (float*)   take((size_t)NNODES * EMB * 4);
  float*    deg  = (float*)   take((size_t)NNODES * 4);
  unsigned* hist = (unsigned*)take(65536 * 4);
  unsigned* cdf  = (unsigned*)take(65536 * 4);
  unsigned* bsum = (unsigned*)take(256 * 4);
  double*   pairSum = (double*)take(64);
  float*    lsums   = (float*) take(64);
  long zeroWords = (long)((off - zeroOff) / 4);

  float* pos     = (float*)d_out;          // d_out[0..E)
  float* outScal = pos + NEDGES;           // d_out[E]=loss, d_out[E+1]=auc

  k_zero   <<<4096, 256, 0, stream>>>((unsigned*)(ws + zeroOff), zeroWords);
  k_prep_x <<<25000, 256, 0, stream>>>(emb, ebias, x1h);
  k_prep_w1<<<320,   256, 0, stream>>>(basis1, root1, Wt1);
  k_prep_w2<<<256,   256, 0, stream>>>(basis2, root2, Wt2);

  // layer 1: [50000x128]f16 x [640x128]f16^T -> [50000x640]f32
  // 3125 M-tiles x 10 N-groups = 31250 wave-strips (16x64 each)
  {
    int waves = 3125 * (640 / 64);
    int blocks = (waves * 32 + 255) / 256;
    k_wmma_gemm<640><<<blocks, 256, 0, stream>>>(x1h, Wt1, xb1, waves);
  }
  k_msg1<<<3200, 256, 0, stream>>>(src, dst, et, comp1, xb1, accA, deg);
  k_fin1<<<25000, 256, 0, stream>>>(accA, deg, xb1, bias1, h1h);

  // layer 2: [50000x128]f16 x [512x128]f16^T -> [50000x512]f32
  // 3125 M-tiles x 8 N-groups = 25000 wave-strips
  {
    int waves = 3125 * (512 / 64);
    int blocks = (waves * 32 + 255) / 256;
    k_wmma_gemm<512><<<blocks, 256, 0, stream>>>(h1h, Wt2, xb2, waves);
  }
  k_msg2<<<3200, 256, 0, stream>>>(src, dst, et, comp2, xb2, accZ);
  k_fin2<<<19532, 256, 0, stream>>>(accZ, deg, xb2, bias2, z);

  k_distmult<<<3200, 256, 0, stream>>>(src, dst, et, negt, z, rel, pos, negb);
  k_loss<<<2048, 256, 0, stream>>>(pos, negb, lsums);

  k_hist <<<2048, 256, 0, stream>>>(negb, hist);
  k_scan1<<<256, 256, 0, stream>>>(hist, cdf, bsum);
  k_scan2<<<1,   256, 0, stream>>>(bsum);
  k_scan3<<<256, 256, 0, stream>>>(cdf, bsum);
  k_pairs<<<2048, 256, 0, stream>>>(pos, hist, cdf, pairSum);
  k_final<<<1, 32, 0, stream>>>(lsums, pairSum, outScal);

  (void)in_sizes; (void)n_in; (void)out_size; (void)ws_size;
}
